// SE3Net_24799141167687
// MI455X (gfx1250) — matrix operand order repaired and checked by reading, not compiled, not run on hardware
//
#include <hip/hip_runtime.h>
#include <hip/hip_bf16.h>
#include <math.h>

typedef _Float16 h16;
typedef __attribute__((ext_vector_type(16))) _Float16 v16h;
typedef __attribute__((ext_vector_type(8)))  float    v8f;

#define NNODES   10000
#define NGRAPHS  16
#define NEDGES   100000
#define EC       12800   // edge chunk (multiple of 16; 100000 = 7*12800 + 10400)

static inline int ceil_div(int a, int b) { return (a + b - 1) / b; }

// ---------------------------------------------------------------------------
// WMMA GEMM:  C[M,N] = act(A[M,Kp] * W[Kp,Np] + bias)
// A: f16 row-major, Kp in {32,128,192}.  W pre-swizzled into B-fragment order:
//   [Kp/32][Np/16][lane 0..31][half 0..15], element (k,n) at
//   k = kt*32 + (lane>=16 ? 16:0) + q ; n = nt*16 + (lane&15)
// Block = 64 threads (2 waves) owning one 16-row M-tile.  The whole 16xKp A
// block is staged to LDS ONCE via GLOBAL_LOAD_ASYNC_TO_LDS_B128 (ASYNCcnt),
// then both waves sweep all N-tiles (nt += 2), re-using the LDS A fragments.
// A/B fragments use contiguous k = 16*(lane/16) + s; C layout per CDNA5 ISA 7.12.2.
// ---------------------------------------------------------------------------
template<int ACT, bool OUTH>
__global__ __launch_bounds__(64) void wmma_gemm(
    const h16* __restrict__ A, const h16* __restrict__ B,
    const float* __restrict__ bias, float* __restrict__ Cf, h16* __restrict__ Ch,
    int M, int Kp, int Np, int Nvalid, int ldc)
{
    __shared__ h16 As[16 * 192];          // max Kp = 192 -> 6KB
    const int lane = threadIdx.x & 31;
    const int wave = threadIdx.x >> 5;
    const int ntiles = Np >> 4;
    const int ksteps = Kp >> 5;
    const int row0 = blockIdx.x << 4;

    // ---- async global->LDS staging of A[16, Kp] (contiguous, lda == Kp) ----
    {
        unsigned ldsbase = (unsigned)(size_t)(void*)As;
        int chunks = Kp * 2;              // #16B chunks in 16*Kp halves (mult of 64)
        for (int c = threadIdx.x; c < chunks; c += 64) {
            unsigned loff = ldsbase + (unsigned)c * 16u;
            const void* g = (const void*)(A + (size_t)row0 * Kp + (size_t)c * 8);
            asm volatile("global_load_async_to_lds_b128 %0, %1, off"
                         :: "v"(loff), "v"(g) : "memory");
        }
        asm volatile("s_wait_asynccnt 0" ::: "memory");
    }
    __syncthreads();

    const int rrow = lane & 15;
    const int hs = lane >> 4;             // 0: K0-15, 1: K16-31
    const int n_lo = lane & 15;
    const int mb = row0 + ((lane >> 4) << 3);   // C layout: vgpr r -> M=r / r+8

    for (int nt = wave; nt < ntiles; nt += 2) {
        union { v8f v; float f[8]; } acc;
        for (int i = 0; i < 8; ++i) acc.f[i] = 0.f;
        for (int kt = 0; kt < ksteps; ++kt) {
            union { v16h v; float4 f4[2]; } a;
            const float4* Ar = (const float4*)(As + rrow * Kp + (kt << 5));
            a.f4[0] = Ar[2 * hs];
            a.f4[1] = Ar[2 * hs + 1];
            const v16h* Bv = (const v16h*)B;
            v16h b = Bv[((size_t)kt * ntiles + nt) * 32 + lane];
            acc.v = __builtin_amdgcn_wmma_f32_16x16x32_f16(
                false, a.v, false, b, (short)0, acc.v, false, false);
        }
        const int n = (nt << 4) + n_lo;
        if (n < Nvalid) {
            const float bv = bias ? bias[n] : 0.f;
            for (int rr = 0; rr < 8; ++rr) {
                float v = acc.f[rr] + bv;
                if (ACT) v = fmaxf(v, 0.f);
                if (OUTH) Ch[(size_t)(mb + rr) * ldc + n] = (h16)v;
                else      Cf[(size_t)(mb + rr) * ldc + n] = v;
            }
        }
    }
}

// Swizzle f32 weight [K,N] (row-major) into WMMA B-fragment layout (f16, padded)
__global__ void k_swizzle_w(const float* __restrict__ W, h16* __restrict__ out,
                            int K, int N, int Kp, int Np)
{
    int idx = blockIdx.x * blockDim.x + threadIdx.x;
    int total = Kp * Np;
    if (idx >= total) return;
    int q = idx & 15; int l2 = idx >> 4;
    int lane = l2 & 31; int l3 = l2 >> 5;
    int ntl = Np >> 4;
    int nt = l3 % ntl, kt = l3 / ntl;
    int k = kt * 32 + ((lane >> 4) << 4) + q;
    int n = nt * 16 + (lane & 15);
    float v = (k < K && n < N) ? W[(size_t)k * N + n] : 0.f;
    out[idx] = (h16)v;
}

// ---------------------------------------------------------------------------
// packing / fills
// ---------------------------------------------------------------------------
__global__ void k_fill(float* p, float v, int n)
{ int t = blockIdx.x * blockDim.x + threadIdx.x; if (t < n) p[t] = v; }

__global__ void k_pack_xrad(const float* __restrict__ ef, const float* __restrict__ rr,
                            h16* __restrict__ out)
{
    int t = blockIdx.x * blockDim.x + threadIdx.x;
    if (t >= NEDGES * 32) return;
    int j = t & 31, e = t >> 5;
    float v = (j < 4) ? ef[e * 4 + j] : (j == 4 ? rr[e] : 0.f);
    out[t] = (h16)v;
}

__global__ void k_pack_f16(const float* __restrict__ x, h16* __restrict__ out,
                           int M, int C, int Kp)
{
    int t = blockIdx.x * blockDim.x + threadIdx.x;
    if (t >= M * Kp) return;
    int j = t % Kp, m = t / Kp;
    out[t] = (h16)((j < C) ? x[(size_t)m * C + j] : 0.f);
}

__global__ void k_concat_pack(const float* __restrict__ a, int ca,
                              const float* __restrict__ b, int cb,
                              h16* __restrict__ out, int M, int Kp)
{
    int t = blockIdx.x * blockDim.x + threadIdx.x;
    if (t >= M * Kp) return;
    int j = t % Kp, m = t / Kp;
    float v = 0.f;
    if (j < ca) v = a[(size_t)m * ca + j];
    else if (j < ca + cb) v = b[(size_t)m * cb + (j - ca)];
    out[t] = (h16)v;
}

// ---------------------------------------------------------------------------
// per-(di,do) tensor-product message: acc[e,o,n] += sum_{f,i,m} R[f,o,i]B[n,m,f]F[i,m]
// basis streamed (prefetched), node features L2-resident gathers
// ---------------------------------------------------------------------------
__global__ void k_pair_msg(const float* __restrict__ R, const float* __restrict__ B,
                           const float* __restrict__ feat, const int* __restrict__ src,
                           float* __restrict__ acc, int e0, int Ec,
                           int nf, int mo, int mi, int ni, int no)
{
    int t = blockIdx.x * blockDim.x + threadIdx.x;
    if (t >= Ec * mo) return;
    int o = t % mo, el = t / mo, e = e0 + el;
    const float* Re = R + (size_t)el * nf * mo * mi;
    const float* Be = B + (size_t)e * no * ni * nf;
    const float* Fe = feat + (size_t)src[e] * mi * ni;
    __builtin_prefetch(Be + (size_t)no * ni * nf * 8, 0, 1);
    float out[5];
    for (int n = 0; n < no; ++n) out[n] = 0.f;
    for (int f = 0; f < nf; ++f)
        for (int i = 0; i < mi; ++i) {
            float rv = Re[((size_t)f * mo + o) * mi + i];
            for (int m = 0; m < ni; ++m) {
                float c = rv * Fe[i * ni + m];
                for (int n = 0; n < no; ++n)
                    out[n] += c * Be[((size_t)n * ni + m) * nf + f];
            }
        }
    float* ac = acc + ((size_t)el * mo + o) * no;
    for (int n = 0; n < no; ++n) ac[n] += out[n];
}

// generic per-node matvec: out[n,o,m] (+)= sum_c W[o,c]*x[n,c,m]
__global__ void k_qproj(const float* __restrict__ W, const float* __restrict__ x,
                        float* __restrict__ out, int N, int O, int C, int md, int accum)
{
    int t = blockIdx.x * blockDim.x + threadIdx.x;
    if (t >= N * O * md) return;
    int m = t % md; int r2 = t / md; int o = r2 % O; int n = r2 / O;
    float s = 0.f;
    for (int c = 0; c < C; ++c) s += W[o * C + c] * x[((size_t)n * C + c) * md + m];
    size_t oi = ((size_t)n * O + o) * md + m;
    if (accum) out[oi] += s; else out[oi] = s;
}

// projection with concat(out_attn[4], h[cinH])
__global__ void k_proj_cat(const float* __restrict__ W, const float* __restrict__ vin,
                           const float* __restrict__ hin, float* __restrict__ out,
                           int N, int cinH, int md)
{
    int t = blockIdx.x * blockDim.x + threadIdx.x;
    if (t >= N * 16 * md) return;
    int m = t % md; int r2 = t / md; int o = r2 % 16; int n = r2 / 16;
    int CC = 4 + cinH;
    float s = 0.f;
    for (int c = 0; c < 4; ++c) s += W[o * CC + c] * vin[((size_t)n * 4 + c) * md + m];
    for (int c = 0; c < cinH; ++c) s += W[o * CC + 4 + c] * hin[((size_t)n * cinH + c) * md + m];
    out[((size_t)n * 16 + o) * md + m] = s;
}

// equivariant norm nonlinearity (in-place), C_MID = 16
__global__ void k_gnorm(float* __restrict__ x, const float* __restrict__ g,
                        const float* __restrict__ be, const float* __restrict__ w,
                        const float* __restrict__ b, int N, int md)
{
    int n = blockIdx.x * blockDim.x + threadIdx.x;
    if (n >= N) return;
    float nr[16]; float mu = 0.f;
    for (int c = 0; c < 16; ++c) {
        float s = 1e-12f;
        for (int m = 0; m < md; ++m) { float v = x[((size_t)n * 16 + c) * md + m]; s += v * v; }
        nr[c] = sqrtf(s); mu += nr[c];
    }
    mu *= (1.f / 16.f);
    float var = 0.f;
    for (int c = 0; c < 16; ++c) { float d = nr[c] - mu; var += d * d; }
    var *= (1.f / 16.f);
    float inv = rsqrtf(var + 1e-5f);
    float ln[16];
    for (int c = 0; c < 16; ++c) ln[c] = (nr[c] - mu) * inv * g[c] + be[c];
    for (int c = 0; c < 16; ++c) {
        float s = b[c];
        for (int cc = 0; cc < 16; ++cc) s += ln[cc] * w[cc * 16 + c];
        s = fmaxf(s, 0.f);
        float sc = s / nr[c];
        for (int m = 0; m < md; ++m) {
            size_t idx = ((size_t)n * 16 + c) * md + m;
            x[idx] = sc * x[idx];
        }
    }
}

// ---------------------------------------------------------------------------
// attention: logits, segment max/sum, scatter
// ---------------------------------------------------------------------------
__device__ __forceinline__ void atomicMaxF(float* addr, float v)
{
    unsigned* ua = (unsigned*)addr;
    unsigned old = *ua;
    while (true) {
        float of = __uint_as_float(old);
        if (of >= v) break;
        unsigned assumed = old;
        old = atomicCAS(ua, assumed, __float_as_uint(v));
        if (old == assumed) break;
    }
}

__global__ void k_logits(const float* k0, const float* q0, const float* k1, const float* q1,
                         const float* k2, const float* q2, const int* __restrict__ dst,
                         float* __restrict__ out, float scale)
{
    int t = blockIdx.x * blockDim.x + threadIdx.x;
    if (t >= NEDGES * 4) return;
    int h = t & 3, e = t >> 2, d = dst[e];
    float s = 0.f;
    if (k0) s += k0[t] * q0[d * 4 + h];
    if (k1) for (int m = 0; m < 3; ++m) s += k1[t * 3 + m] * q1[(d * 4 + h) * 3 + m];
    if (k2) for (int m = 0; m < 5; ++m) s += k2[t * 5 + m] * q2[(d * 4 + h) * 5 + m];
    out[t] = s * scale;
}

__global__ void k_segmax(const float* __restrict__ lg, const int* __restrict__ dst,
                         float* __restrict__ emax)
{
    int t = blockIdx.x * blockDim.x + threadIdx.x;
    if (t >= NEDGES * 4) return;
    atomicMaxF(&emax[dst[t >> 2] * 4 + (t & 3)], lg[t]);
}

__global__ void k_expsum(float* __restrict__ lg, const float* __restrict__ emax,
                         const int* __restrict__ dst, float* __restrict__ z)
{
    int t = blockIdx.x * blockDim.x + threadIdx.x;
    if (t >= NEDGES * 4) return;
    int h = t & 3, e = t >> 2;
    float ee = expf(lg[t] - emax[dst[e] * 4 + h]);
    lg[t] = ee;
    atomicAdd(&z[dst[e] * 4 + h], ee);
}

__global__ void k_scatter_attn(const float* __restrict__ ee, const float* __restrict__ z,
                               const int* __restrict__ dst,
                               const float* v0, const float* v1, const float* v2,
                               float* o0, float* o1, float* o2)
{
    int t = blockIdx.x * blockDim.x + threadIdx.x;
    if (t >= NEDGES * 4) return;
    int h = t & 3, e = t >> 2, d = dst[e];
    float a = ee[t] / (z[d * 4 + h] + 1e-9f);
    atomicAdd(&o0[d * 4 + h], a * v0[t]);
    for (int m = 0; m < 3; ++m) atomicAdd(&o1[(d * 4 + h) * 3 + m], a * v1[t * 3 + m]);
    for (int m = 0; m < 5; ++m) atomicAdd(&o2[(d * 4 + h) * 5 + m], a * v2[t * 5 + m]);
}

// ---------------------------------------------------------------------------
// head scatter / pooling / final
// ---------------------------------------------------------------------------
__global__ void k_scatter_head(const float* __restrict__ acc, const int* __restrict__ dst,
                               float* __restrict__ nodeOut, int e0, int Ec)
{
    int t = blockIdx.x * blockDim.x + threadIdx.x;
    if (t >= Ec * 48) return;
    int o = t % 48, el = t / 48;
    atomicAdd(&nodeOut[(size_t)dst[e0 + el] * 48 + o], acc[(size_t)el * 48 + o]);
}

__global__ void k_count(const int* __restrict__ gid, float* __restrict__ cnt)
{
    int n = blockIdx.x * blockDim.x + threadIdx.x;
    if (n < NNODES) atomicAdd(&cnt[gid[n]], 1.f);
}

__global__ void k_pool(const float* __restrict__ nodeOut, const int* __restrict__ gid,
                       float* __restrict__ pool)
{
    int t = blockIdx.x * blockDim.x + threadIdx.x;
    if (t >= NNODES * 48) return;
    int o = t % 48, n = t / 48;
    atomicAdd(&pool[gid[n] * 48 + o], nodeOut[t]);
}

__global__ void k_divide(float* pool, const float* cnt)
{
    int t = blockIdx.x * blockDim.x + threadIdx.x;
    if (t < NGRAPHS * 48) pool[t] /= cnt[t / 48];
}

__global__ void k_sigmoid2(const float* __restrict__ y, float* __restrict__ xs,
                           float* __restrict__ dout, int n)
{
    int t = blockIdx.x * blockDim.x + threadIdx.x;
    if (t >= n) return;
    float v = 1.f / (1.f + expf(-y[t]));
    xs[t] = v; dout[t] = v;
}

// ---------------------------------------------------------------------------
// host orchestration
// ---------------------------------------------------------------------------
struct Ws { char* base; size_t off, cap; };
static void* wsalloc(Ws& w, size_t bytes)
{ size_t a = (w.off + 255) & ~(size_t)255; w.off = a + bytes; return w.base + a; }

static h16* swz(Ws& ws, hipStream_t s, const float* W, int K, int N, int* KpO, int* NpO)
{
    int Kp = ceil_div(K, 32) * 32, Np = ceil_div(N, 16) * 16;
    h16* out = (h16*)wsalloc(ws, (size_t)Kp * Np * sizeof(h16));
    int total = Kp * Np;
    k_swizzle_w<<<ceil_div(total, 256), 256, 0, s>>>(W, out, K, N, Kp, Np);
    if (KpO) *KpO = Kp; if (NpO) *NpO = Np;
    return out;
}

struct Rad { const h16 *w1, *w2, *w3; const float *b1, *b2, *b3; int nout; };
static Rad prep_rad(Ws& ws, hipStream_t s, void* const* din, int base, int nout)
{
    Rad r;
    r.b1 = (const float*)din[base + 0];
    r.b2 = (const float*)din[base + 1];
    r.b3 = (const float*)din[base + 2];
    r.w1 = swz(ws, s, (const float*)din[base + 3], 5, 32, nullptr, nullptr);
    r.w2 = swz(ws, s, (const float*)din[base + 4], 32, 32, nullptr, nullptr);
    r.w3 = swz(ws, s, (const float*)din[base + 5], 32, nout, nullptr, nullptr);
    r.nout = nout;
    return r;
}

struct Lin { const h16 *w1, *w2; const float *b1, *b2; int k1p, n1p, k2p, n2p; };
static Lin prep_lin(Ws& ws, hipStream_t s, void* const* din, int base, int nin, int nh, int nout)
{
    Lin l;
    l.b1 = (const float*)din[base + 0];
    l.b2 = (const float*)din[base + 1];
    l.w1 = swz(ws, s, (const float*)din[base + 2], nin, nh, &l.k1p, &l.n1p);
    l.w2 = swz(ws, s, (const float*)din[base + 3], nh, nout, &l.k2p, &l.n2p);
    return l;
}

static void gemm(hipStream_t s, const h16* A, const h16* B, const float* bias,
                 int M, int Kp, int Np, int Nvalid, int act, h16* oh, float* of, int ldc)
{
    dim3 g(M / 16), b(64);
    if (oh) {
        if (act) wmma_gemm<1, true ><<<g, b, 0, s>>>(A, B, bias, nullptr, oh, M, Kp, Np, Nvalid, ldc);
        else     wmma_gemm<0, true ><<<g, b, 0, s>>>(A, B, bias, nullptr, oh, M, Kp, Np, Nvalid, ldc);
    } else {
        if (act) wmma_gemm<1, false><<<g, b, 0, s>>>(A, B, bias, of, nullptr, M, Kp, Np, Nvalid, ldc);
        else     wmma_gemm<0, false><<<g, b, 0, s>>>(A, B, bias, of, nullptr, M, Kp, Np, Nvalid, ldc);
    }
}

static void run_mlp3(hipStream_t s, const Rad& r, const h16* X, int Ec, h16* H1, h16* H2, float* R)
{
    gemm(s, X,  r.w1, r.b1, Ec, 32, 32, 32, 1, H1, nullptr, 32);
    gemm(s, H1, r.w2, r.b2, Ec, 32, 32, 32, 1, H2, nullptr, 32);
    gemm(s, H2, r.w3, r.b3, Ec, 32, r.nout, r.nout, 0, nullptr, R, r.nout);
}

static void fillf(hipStream_t s, float* p, float v, int n)
{ k_fill<<<ceil_div(n, 256), 256, 0, s>>>(p, v, n); }

extern "C" void kernel_launch(void* const* d_in, const int* in_sizes, int n_in,
                              void* d_out, int out_size, void* d_ws, size_t ws_size,
                              hipStream_t stream)
{
    (void)in_sizes; (void)out_size;
    if (n_in < 236) return; // pytree flatten: 4 + 220 params + 9 basis + 3
    hipStream_t s = stream;

    const float* node_feat = (const float*)d_in[0];   // [N,32,1]
    const float* edge_feat = (const float*)d_in[1];   // [E,4]
    const float* r_in      = (const float*)d_in[2];   // [E,1]
    const float* ij        = (const float*)d_in[3];   // [16,21]
    const int*   src       = (const int*)d_in[233];
    const int*   dst       = (const int*)d_in[234];
    const int*   gid       = (const int*)d_in[235];
    const float* basis[9];
    for (int i = 0; i < 9; ++i) basis[i] = (const float*)d_in[224 + i]; // '0,0'..'2,2'

    Ws ws{(char*)d_ws, 0, ws_size};

    // ---------------- weight prep (sorted-key pytree offsets) ----------------
    Rad l0v[3], l0k, l1v[9], l1k[9], hreg[3], hcls[3];
    for (int dd = 0; dd < 3; ++dd) l0v[dd] = prep_rad(ws, s, d_in, 45 + 6 * dd, 128);
    l0k = prep_rad(ws, s, d_in, 23, 128);
    for (int p = 0; p < 9; ++p) {
        int di = p / 3, dd = p % 3;
        int nf = 2 * (di < dd ? di : dd) + 1;
        l1k[p] = prep_rad(ws, s, d_in, 63 + 6 * p, nf * 64);
        l1v[p] = prep_rad(ws, s, d_in, 135 + 6 * p, nf * 64);
    }
    for (int di = 0; di < 3; ++di) {
        hcls[di] = prep_rad(ws, s, d_in, 4 + 6 * di, 768);
        hreg[di] = prep_rad(ws, s, d_in, 205 + 6 * di, 768);
    }
    Lin lembC  = prep_lin(ws, s, d_in, 189, 21, 128, 128);
    Lin lheadC = prep_lin(ws, s, d_in, 193, 176, 176, 21);
    Lin lembR  = prep_lin(ws, s, d_in, 197, 21, 128, 128);
    Lin lheadR = prep_lin(ws, s, d_in, 201, 176, 176, 1);

    // ---------------- scratch buffers ----------------
    h16*   Xrad = (h16*)wsalloc(ws, (size_t)NEDGES * 32 * sizeof(h16));
    h16*   H1   = (h16*)wsalloc(ws, (size_t)EC * 32 * sizeof(h16));
    h16*   H2   = (h16*)wsalloc(ws, (size_t)EC * 32 * sizeof(h16));
    float* Rbuf = (float*)wsalloc(ws, (size_t)EC * 768 * sizeof(float));
    float* vacc[3], *kacc[3], *qb[3], *ob[3], *h1b[3], *h2b[3];
    for (int d = 0; d < 3; ++d) {
        int md = 2 * d + 1;
        vacc[d] = (float*)wsalloc(ws, (size_t)NEDGES * 4 * md * sizeof(float));
        kacc[d] = (float*)wsalloc(ws, (size_t)NEDGES * 4 * md * sizeof(float));
        qb[d]   = (float*)wsalloc(ws, (size_t)NNODES * 4 * md * sizeof(float));
        ob[d]   = (float*)wsalloc(ws, (size_t)NNODES * 4 * md * sizeof(float));
        h1b[d]  = (float*)wsalloc(ws, (size_t)NNODES * 16 * md * sizeof(float));
        h2b[d]  = (float*)wsalloc(ws, (size_t)NNODES * 16 * md * sizeof(float));
    }
    float* logit   = (float*)wsalloc(ws, (size_t)NEDGES * 4 * sizeof(float));
    float* emax    = (float*)wsalloc(ws, (size_t)NNODES * 4 * sizeof(float));
    float* zb      = (float*)wsalloc(ws, (size_t)NNODES * 4 * sizeof(float));
    float* accHead = (float*)wsalloc(ws, (size_t)EC * 48 * sizeof(float));
    float* nodeOut = (float*)wsalloc(ws, (size_t)NNODES * 48 * sizeof(float));
    float* poolR   = (float*)wsalloc(ws, 16 * 48 * sizeof(float));
    float* poolC   = (float*)wsalloc(ws, 16 * 48 * sizeof(float));
    float* cnt     = (float*)wsalloc(ws, 16 * sizeof(float));
    float* embf    = (float*)wsalloc(ws, 16 * 128 * sizeof(float));
    h16*   sm16a   = (h16*)wsalloc(ws, 16 * 192 * sizeof(h16));
    h16*   sm16b   = (h16*)wsalloc(ws, 16 * 192 * sizeof(h16));
    float* ybuf    = (float*)wsalloc(ws, 16 * 21 * sizeof(float));
    float* xsbuf   = (float*)wsalloc(ws, 16 * 21 * sizeof(float));

    // ---------------- x_rad = [edge_feat | r | 0-pad] as f16 ----------------
    k_pack_xrad<<<ceil_div(NEDGES * 32, 256), 256, 0, s>>>(edge_feat, r_in, Xrad);

    const int E4 = NEDGES * 4, TB = 256;

    // =========================== LAYER 0 (in_degs={0}) ===========================
    for (int d = 0; d < 3; ++d) fillf(s, vacc[d], 0.f, NEDGES * 4 * (2 * d + 1));
    fillf(s, kacc[0], 0.f, NEDGES * 4);
    for (int e0 = 0; e0 < NEDGES; e0 += EC) {
        int Ec = (NEDGES - e0 < EC) ? (NEDGES - e0) : EC;
        for (int dd = 0; dd < 3; ++dd) {
            int no = 2 * dd + 1;
            run_mlp3(s, l0v[dd], Xrad + (size_t)e0 * 32, Ec, H1, H2, Rbuf);
            k_pair_msg<<<ceil_div(Ec * 4, TB), TB, 0, s>>>(Rbuf, basis[dd], node_feat, src,
                vacc[dd] + (size_t)e0 * 4 * no, e0, Ec, 1, 4, 32, 1, no);
        }
        run_mlp3(s, l0k, Xrad + (size_t)e0 * 32, Ec, H1, H2, Rbuf);
        k_pair_msg<<<ceil_div(Ec * 4, TB), TB, 0, s>>>(Rbuf, basis[0], node_feat, src,
            kacc[0] + (size_t)e0 * 4, e0, Ec, 1, 4, 32, 1, 1);
    }
    k_qproj<<<ceil_div(NNODES * 4, TB), TB, 0, s>>>((const float*)d_in[44], node_feat,
                                                    qb[0], NNODES, 4, 32, 1, 0);
    // attention (dk = 1)
    fillf(s, emax, -1e30f, NNODES * 4); fillf(s, zb, 0.f, NNODES * 4);
    k_logits<<<ceil_div(E4, TB), TB, 0, s>>>(kacc[0], qb[0], nullptr, nullptr, nullptr, nullptr,
                                             dst, logit, 1.0f);
    k_segmax<<<ceil_div(E4, TB), TB, 0, s>>>(logit, dst, emax);
    k_expsum<<<ceil_div(E4, TB), TB, 0, s>>>(logit, emax, dst, zb);
    for (int d = 0; d < 3; ++d) fillf(s, ob[d], 0.f, NNODES * 4 * (2 * d + 1));
    k_scatter_attn<<<ceil_div(E4, TB), TB, 0, s>>>(logit, zb, dst, vacc[0], vacc[1], vacc[2],
                                                   ob[0], ob[1], ob[2]);
    // projection + norm
    k_proj_cat<<<ceil_div(NNODES * 16 * 1, TB), TB, 0, s>>>((const float*)d_in[41], ob[0],
                                                            node_feat, h1b[0], NNODES, 32, 1);
    k_proj_cat<<<ceil_div(NNODES * 16 * 3, TB), TB, 0, s>>>((const float*)d_in[42], ob[1],
                                                            nullptr, h1b[1], NNODES, 0, 3);
    k_proj_cat<<<ceil_div(NNODES * 16 * 5, TB), TB, 0, s>>>((const float*)d_in[43], ob[2],
                                                            nullptr, h1b[2], NNODES, 0, 5);
    for (int d = 0; d < 3; ++d) {
        int nb = 29 + 4 * d; // b, beta, gamma, w
        k_gnorm<<<ceil_div(NNODES, TB), TB, 0, s>>>(h1b[d], (const float*)d_in[nb + 2],
            (const float*)d_in[nb + 1], (const float*)d_in[nb + 3], (const float*)d_in[nb + 0],
            NNODES, 2 * d + 1);
    }

    // =========================== LAYER 1 (in_degs={0,1,2}) ===========================
    for (int d = 0; d < 3; ++d) {
        fillf(s, vacc[d], 0.f, NEDGES * 4 * (2 * d + 1));
        fillf(s, kacc[d], 0.f, NEDGES * 4 * (2 * d + 1));
    }
    for (int e0 = 0; e0 < NEDGES; e0 += EC) {
        int Ec = (NEDGES - e0 < EC) ? (NEDGES - e0) : EC;
        for (int p = 0; p < 9; ++p) {
            int di = p / 3, dd = p % 3;
            int nf = 2 * (di < dd ? di : dd) + 1;
            int ni = 2 * di + 1, no = 2 * dd + 1;
            run_mlp3(s, l1v[p], Xrad + (size_t)e0 * 32, Ec, H1, H2, Rbuf);
            k_pair_msg<<<ceil_div(Ec * 4, TB), TB, 0, s>>>(Rbuf, basis[p], h1b[di], src,
                vacc[dd] + (size_t)e0 * 4 * no, e0, Ec, nf, 4, 16, ni, no);
            run_mlp3(s, l1k[p], Xrad + (size_t)e0 * 32, Ec, H1, H2, Rbuf);
            k_pair_msg<<<ceil_div(Ec * 4, TB), TB, 0, s>>>(Rbuf, basis[p], h1b[di], src,
                kacc[dd] + (size_t)e0 * 4 * no, e0, Ec, nf, 4, 16, ni, no);
        }
    }
    for (int d = 0; d < 3; ++d)
        k_qproj<<<ceil_div(NNODES * 4 * (2 * d + 1), TB), TB, 0, s>>>((const float*)d_in[132 + d],
            h1b[d], qb[d], NNODES, 4, 16, 2 * d + 1, 0);
    // attention (dk = 9)
    fillf(s, emax, -1e30f, NNODES * 4); fillf(s, zb, 0.f, NNODES * 4);
    k_logits<<<ceil_div(E4, TB), TB, 0, s>>>(kacc[0], qb[0], kacc[1], qb[1], kacc[2], qb[2],
                                             dst, logit, 1.0f / 3.0f);
    k_segmax<<<ceil_div(E4, TB), TB, 0, s>>>(logit, dst, emax);
    k_expsum<<<ceil_div(E4, TB), TB, 0, s>>>(logit, emax, dst, zb);
    for (int d = 0; d < 3; ++d) fillf(s, ob[d], 0.f, NNODES * 4 * (2 * d + 1));
    k_scatter_attn<<<ceil_div(E4, TB), TB, 0, s>>>(logit, zb, dst, vacc[0], vacc[1], vacc[2],
                                                   ob[0], ob[1], ob[2]);
    for (int d = 0; d < 3; ++d) {
        int md = 2 * d + 1;
        k_proj_cat<<<ceil_div(NNODES * 16 * md, TB), TB, 0, s>>>((const float*)d_in[129 + d],
            ob[d], h1b[d], h2b[d], NNODES, 16, md);
        int nb = 117 + 4 * d;
        k_gnorm<<<ceil_div(NNODES, TB), TB, 0, s>>>(h2b[d], (const float*)d_in[nb + 2],
            (const float*)d_in[nb + 1], (const float*)d_in[nb + 3], (const float*)d_in[nb + 0],
            NNODES, md);
    }

    // =========================== OUTPUT HEADS ===========================
    fillf(s, cnt, 0.f, 16);
    k_count<<<ceil_div(NNODES, TB), TB, 0, s>>>(gid, cnt);
    for (int hh = 0; hh < 2; ++hh) {
        const Rad* hr = hh ? hcls : hreg;
        const float* selfW = (const float*)d_in[hh ? 22 : 223];
        float* pool = hh ? poolC : poolR;
        fillf(s, nodeOut, 0.f, NNODES * 48);
        for (int e0 = 0; e0 < NEDGES; e0 += EC) {
            int Ec = (NEDGES - e0 < EC) ? (NEDGES - e0) : EC;
            fillf(s, accHead, 0.f, Ec * 48);
            for (int di = 0; di < 3; ++di) {
                run_mlp3(s, hr[di], Xrad + (size_t)e0 * 32, Ec, H1, H2, Rbuf);
                k_pair_msg<<<ceil_div(Ec * 48, TB), TB, 0, s>>>(Rbuf, basis[di * 3], h2b[di],
                    src, accHead, e0, Ec, 1, 48, 16, 2 * di + 1, 1);
            }
            k_scatter_head<<<ceil_div(Ec * 48, TB), TB, 0, s>>>(accHead, dst, nodeOut, e0, Ec);
        }
        k_qproj<<<ceil_div(NNODES * 48, TB), TB, 0, s>>>(selfW, h2b[0], nodeOut,
                                                         NNODES, 48, 16, 1, 1);
        fillf(s, pool, 0.f, 16 * 48);
        k_pool<<<ceil_div(NNODES * 48, TB), TB, 0, s>>>(nodeOut, gid, pool);
        k_divide<<<ceil_div(16 * 48, TB), TB, 0, s>>>(pool, cnt);
    }

    // =========================== FINAL MLPs (M=16 WMMA tiles) ===========================
    // class path: y = head(concat(poolC, emb(ij))) ; xs = sigmoid(y)
    k_pack_f16<<<ceil_div(16 * 32, TB), TB, 0, s>>>(ij, sm16a, 16, 21, 32);
    gemm(s, sm16a, lembC.w1, lembC.b1, 16, 32, 128, 128, 1, sm16b, nullptr, 128);
    gemm(s, sm16b, lembC.w2, lembC.b2, 16, 128, 128, 128, 0, nullptr, embf, 128);
    k_concat_pack<<<ceil_div(16 * 192, TB), TB, 0, s>>>(poolC, 48, embf, 128, sm16a, 16, 192);
    fillf(s, (float*)sm16b, 0.f, 16 * 192 / 2);
    gemm(s, sm16a, lheadC.w1, lheadC.b1, 16, 192, 176, 176, 1, sm16b, nullptr, 192);
    gemm(s, sm16b, lheadC.w2, lheadC.b2, 16, 192, 32, 21, 0, nullptr, ybuf, 21);
    k_sigmoid2<<<ceil_div(16 * 21, TB), TB, 0, s>>>(ybuf, xsbuf, (float*)d_out + 16, 16 * 21);
    // reg path: pred = head(concat(poolR, emb(xs)))
    k_pack_f16<<<ceil_div(16 * 32, TB), TB, 0, s>>>(xsbuf, sm16a, 16, 21, 32);
    gemm(s, sm16a, lembR.w1, lembR.b1, 16, 32, 128, 128, 1, sm16b, nullptr, 128);
    gemm(s, sm16b, lembR.w2, lembR.b2, 16, 128, 128, 128, 0, nullptr, embf, 128);
    k_concat_pack<<<ceil_div(16 * 192, TB), TB, 0, s>>>(poolR, 48, embf, 128, sm16a, 16, 192);
    fillf(s, (float*)sm16b, 0.f, 16 * 192 / 2);
    gemm(s, sm16a, lheadR.w1, lheadR.b1, 16, 192, 176, 176, 1, sm16b, nullptr, 192);
    gemm(s, sm16b, lheadR.w2, lheadR.b2, 16, 192, 16, 1, 0, nullptr, (float*)d_out, 1);
}
